// GeometryAwareDenseFieldViewNet_23553600651602
// MI455X (gfx1250) — matrix operand order repaired and checked by reading, not compiled, not run on hardware
//
#include <hip/hip_runtime.h>
#include <math.h>

// ---- problem constants (match reference) ----
constexpr int cB = 2, cM = 1024, cV = 300, cC = 128, cHID = 128;
constexpr int cBIN = 256, cH = 384, cW = 512;
constexpr int cR  = cB * cM;   // 2048 token rows
constexpr int cKQ = 160;       // q_in K padded: 128+7 -> 160 (multiple of 32)
constexpr int cVP = 304;       // logits N padded: 300 -> 304 (multiple of 16)

typedef __attribute__((ext_vector_type(16))) __bf16 v16bf;
typedef __attribute__((ext_vector_type(8)))  float  v8f;
typedef __attribute__((ext_vector_type(4)))  unsigned int u32x4;

__device__ __forceinline__ unsigned short f2bf(float f) {
  unsigned int u = __float_as_uint(f);
  u += 0x7fffu + ((u >> 16) & 1u);     // round-to-nearest-even
  return (unsigned short)(u >> 16);
}

// Decode packed-B element index -> (k, col) for a buffer with nk k-tiles.
// Packed layout: Bp[(((tn*nk)+kt)*32 + lane)*16 + h], where the wave32 WMMA
// B-fragment order used here (contiguous k, same as A) is: col = tn*16 + (lane&15),
// k = kt*32 + 16*(lane>>4) + h.
__device__ __forceinline__ void unpack_bidx(int i, int nk, int& k, int& col) {
  int h    = i & 15;
  int lane = (i >> 4) & 31;
  int rem  = i >> 9;
  int kt   = rem % nk;
  int tn   = rem / nk;
  col = tn * 16 + (lane & 15);
  k   = kt * 32 + 16 * (lane >> 4) + h;
}

// ---------------------------------------------------------------------------
// K0: f32 weights -> bf16, written directly in WMMA-B fragment-major order.
// ---------------------------------------------------------------------------
__global__ void convert_weights_kernel(const float* __restrict__ Wq1,
                                       const float* __restrict__ Wq2,
                                       const float* __restrict__ Wr1,
                                       const float* __restrict__ Wr2,
                                       unsigned short* Wq1b, unsigned short* Wq2b,
                                       unsigned short* Wr1b, unsigned short* Wr2b) {
  int i = blockIdx.x * blockDim.x + threadIdx.x;
  int k, col;
  if (i < cKQ * cHID) {                       // 8 n-tiles x 5 k-tiles
    unpack_bidx(i, cKQ / 32, k, col);
    float v = (k < cC + 7) ? Wq1[k * cHID + col] : 0.f;
    Wq1b[i] = f2bf(v);
    return;
  }
  i -= cKQ * cHID;
  if (i < cHID * cHID) { unpack_bidx(i, 4, k, col); Wq2b[i] = f2bf(Wq2[k * cHID + col]); return; }
  i -= cHID * cHID;
  if (i < cHID * cC)   { unpack_bidx(i, 4, k, col); Wr1b[i] = f2bf(Wr1[k * cC + col]);  return; }
  i -= cHID * cC;
  if (i < cC * cC)     { unpack_bidx(i, 4, k, col); Wr2b[i] = f2bf(Wr2[k * cC + col]);  return; }
}

// ---------------------------------------------------------------------------
// K1: anchor MLP a=(relu(vd@Wa1+ba1))@Wa2+ba2  and bias=(relu(vd@Wb1+bb1))@Wb2+bb2
// one block per view (300 views) -- tiny, plain VALU
// ---------------------------------------------------------------------------
__global__ __launch_bounds__(128) void anchor_kernel(
    const float* __restrict__ vd,
    const float* __restrict__ Wa1, const float* __restrict__ ba1,
    const float* __restrict__ Wa2, const float* __restrict__ ba2,
    const float* __restrict__ Wb1, const float* __restrict__ bb1,
    const float* __restrict__ Wb2, const float* __restrict__ bb2,
    float* __restrict__ a_out, float* __restrict__ bias_out) {
  __shared__ float hid[cHID];
  __shared__ float hb[cHID / 2];
  int v = blockIdx.x;
  int j = threadIdx.x;
  float d0 = vd[v * 3 + 0], d1 = vd[v * 3 + 1], d2 = vd[v * 3 + 2];
  float h = d0 * Wa1[0 * cHID + j] + d1 * Wa1[1 * cHID + j] + d2 * Wa1[2 * cHID + j] + ba1[j];
  hid[j] = fmaxf(h, 0.f);
  if (j < 64) {
    float g = d0 * Wb1[0 * 64 + j] + d1 * Wb1[1 * 64 + j] + d2 * Wb1[2 * 64 + j] + bb1[j];
    hb[j] = fmaxf(g, 0.f);
  }
  __syncthreads();
  float s = ba2[j];
  #pragma unroll 8
  for (int k = 0; k < cHID; ++k) s += hid[k] * Wa2[k * cHID + j];
  a_out[v * cHID + j] = s;
  if (j == 0) {
    float sb = bb2[0];
    for (int k = 0; k < 64; ++k) sb += hb[k] * Wb2[k];
    bias_out[v] = sb;
  }
}

// ---------------------------------------------------------------------------
// K1b: a(300x128) f32 -> aT bf16 in packed B-fragment order (19 n-tiles, 4 k-tiles)
// ---------------------------------------------------------------------------
__global__ void transpose_a_kernel(const float* __restrict__ a, unsigned short* __restrict__ ab) {
  int i = blockIdx.x * blockDim.x + threadIdx.x;
  if (i >= (cVP / 16) * 4 * 512) return;      // 19*4*512 = 38912
  int k, col;
  unpack_bidx(i, 4, k, col);
  float v = (col < cV) ? a[col * cHID + k] : 0.f;
  ab[i] = f2bf(v);
}

// ---------------------------------------------------------------------------
// K2: per-token geometry -> q_in row (bf16, 160 cols).  One wave32 per token.
// ---------------------------------------------------------------------------
__global__ __launch_bounds__(256) void geom_kernel(
    const int*   __restrict__ tok,
    const float* __restrict__ Kc,
    const float* __restrict__ depth,
    const float* __restrict__ dprob,
    const float* __restrict__ seed,
    unsigned short* __restrict__ qin) {
  int lane = threadIdx.x & 31;
  int t = (int)((blockIdx.x * blockDim.x + threadIdx.x) >> 5);
  if (t >= cR) return;
  int b = t >> 10, m = t & (cM - 1);
  int pix = tok[b * cM + m];
  int px = pix % cW, py = pix / cW;
  float fx = Kc[b * 9 + 0], cx = Kc[b * 9 + 2];
  float fy = Kc[b * 9 + 4], cy = Kc[b * 9 + 5];

  // entropy over 256 bins: 8 bins per lane
  const float* pp = dprob + (size_t)b * cBIN * cH * cW + (size_t)py * cW + px;
  float s = 0.f;
  #pragma unroll
  for (int k = 0; k < 8; ++k) {
    float p = pp[(size_t)(lane * 8 + k) * (cH * cW)];
    p = fmaxf(p, 1e-8f);
    s -= p * logf(p);
  }
  #pragma unroll
  for (int o = 16; o > 0; o >>= 1) s += __shfl_xor(s, o, 32);
  float unc = s * (1.f / 5.545177444479562f);         // / log(256)
  unc = fminf(fmaxf(unc, 0.f), 1.f);

  // pack seed features (all lanes, 4 cols each)
  const float* sf = seed + ((size_t)b * cC) * cM + m;
  unsigned short* qrow = qin + (size_t)t * cKQ;
  #pragma unroll
  for (int k = 0; k < 4; ++k) {
    int c = lane + 32 * k;
    qrow[c] = f2bf(sf[(size_t)c * cM]);
  }
  if (lane < cKQ - (cC + 7)) qrow[cC + 7 + lane] = 0;   // zero pad cols 135..159

  if (lane == 0) {
    float rx = ((float)px - cx) / fx, ry = ((float)py - cy) / fy;
    float rn = sqrtf(rx * rx + ry * ry + 1.f);
    float r0 = rx / rn, r1 = ry / rn, r2 = 1.f / rn;
    int xc = px < 1 ? 1 : (px > cW - 2 ? cW - 2 : px);
    int yc = py < 1 ? 1 : (py > cH - 2 ? cH - 2 : py);
    const float* dm = depth + (size_t)b * cH * cW;
    float zxp = dm[yc * cW + xc + 1], zxm = dm[yc * cW + xc - 1];
    float zyp = dm[(yc + 1) * cW + xc], zym = dm[(yc - 1) * cW + xc];
    float dx0 = ((float)(xc + 1) - cx) / fx * zxp - ((float)(xc - 1) - cx) / fx * zxm;
    float dx1 = ((float)yc - cy) / fy * zxp - ((float)yc - cy) / fy * zxm;
    float dx2 = zxp - zxm;
    float dy0 = ((float)xc - cx) / fx * zyp - ((float)xc - cx) / fx * zym;
    float dy1 = ((float)(yc + 1) - cy) / fy * zyp - ((float)(yc - 1) - cy) / fy * zym;
    float dy2 = zyp - zym;
    float n0 = dx1 * dy2 - dx2 * dy1;
    float n1 = dx2 * dy0 - dx0 * dy2;
    float n2 = dx0 * dy1 - dx1 * dy0;
    float nn = fmaxf(sqrtf(n0 * n0 + n1 * n1 + n2 * n2), 1e-6f);
    float wsc = (1.f - unc) / nn;
    qrow[128] = f2bf(r0);       qrow[129] = f2bf(r1);       qrow[130] = f2bf(r2);
    qrow[131] = f2bf(n0 * wsc); qrow[132] = f2bf(n1 * wsc); qrow[133] = f2bf(n2 * wsc);
    qrow[134] = f2bf(unc);
  }
}

// ---------------------------------------------------------------------------
// WMMA GEMM, K and MODE compile-time so the k-loop fully unrolls.
//   C(2048 x N) = A(2048 x K, row-major bf16) @ Bp(packed fragment-major bf16)
// A fragment: two 16B loads / k-tile; B fragment: two 16B loads (pre-swizzled).
// MODE 1: relu -> outF and/or outB;  MODE 2: sigmoid(acc*scale+bias) -> outF;
// MODE 3: (B,C,M)-transposed store of acc+bias -> outF.
// ---------------------------------------------------------------------------
template <int K, int MODE>
__global__ __launch_bounds__(128) void wmma_gemm_bf16(
    const unsigned short* __restrict__ A, int lda,
    const unsigned short* __restrict__ Bp,
    const float* __restrict__ bias,
    float* __restrict__ outF, unsigned short* __restrict__ outB,
    int N, float scale) {
  constexpr int NK = K / 32;
  int lane = threadIdx.x & 31;
  int wave = threadIdx.x >> 5;
  int tile_m = blockIdx.x;
  int tile_n = blockIdx.y * 4 + wave;
  if (tile_n * 16 >= N) return;            // wave-uniform -> EXEC stays all-ones

  int hi = lane >> 4;
  int r  = lane & 15;
  int rowA = tile_m * 16 + r;

  union { v16bf v; u32x4 q[2]; } ua, ub;
  v8f acc = {0.f, 0.f, 0.f, 0.f, 0.f, 0.f, 0.f, 0.f};

  const unsigned short* ap = A + (size_t)rowA * lda + 16 * hi;
  const unsigned short* bp = Bp + ((size_t)(tile_n * NK) * 32 + lane) * 16;
  #pragma unroll
  for (int kt = 0; kt < NK; ++kt) {
    ua.q[0] = *reinterpret_cast<const u32x4*>(ap + kt * 32);
    ua.q[1] = *reinterpret_cast<const u32x4*>(ap + kt * 32 + 8);
    ub.q[0] = *reinterpret_cast<const u32x4*>(bp + kt * 512);
    ub.q[1] = *reinterpret_cast<const u32x4*>(bp + kt * 512 + 8);
    acc = __builtin_amdgcn_wmma_f32_16x16x32_bf16(false, ua.v, false, ub.v,
                                                  (short)0, acc, false, false);
  }

  int col = tile_n * 16 + r;
  bool colOK = (col < N);
  float bv = colOK ? bias[col] : 0.f;
  #pragma unroll
  for (int g = 0; g < 8; ++g) {
    int row = tile_m * 16 + g + 8 * hi;
    float av = acc[g];
    float o;
    if (MODE == 2) o = 1.f / (1.f + expf(-(av * scale + bv)));
    else { o = av + bv; if (MODE == 1) o = fmaxf(o, 0.f); }
    if (colOK) {
      if (MODE == 3) {
        int b = row >> 10, m = row & (cM - 1);
        outF[((size_t)(b * cC + col)) * cM + m] = o;       // (B,C,M) layout
      } else {
        if (outF) outF[(size_t)row * N + col] = o;
        if (outB) outB[(size_t)row * N + col] = f2bf(o);
      }
    }
  }
}

// ---------------------------------------------------------------------------
// K4: per-token argmax over V=300 scores (first occurrence on ties),
// writes top_inds / top_xyz / rot, and rin = bf16(q + a[top]).
// ---------------------------------------------------------------------------
__global__ __launch_bounds__(256) void select_kernel(
    const float* __restrict__ vscore, const float* __restrict__ vd,
    const float* __restrict__ a_f32, const float* __restrict__ q_f32,
    float* __restrict__ o_ind, float* __restrict__ o_xyz, float* __restrict__ o_rot,
    unsigned short* __restrict__ rin) {
  int lane = threadIdx.x & 31;
  int t = (int)((blockIdx.x * blockDim.x + threadIdx.x) >> 5);
  if (t >= cR) return;
  const float* row = vscore + (size_t)t * cV;
  float best = -1e30f; int bi = cV;
  for (int v = lane; v < cV; v += 32) {
    float s = row[v];
    if (s > best) { best = s; bi = v; }
  }
  #pragma unroll
  for (int o = 16; o > 0; o >>= 1) {
    float ob = __shfl_xor(best, o, 32);
    int   oi = __shfl_xor(bi, o, 32);
    if (ob > best || (ob == best && oi < bi)) { best = ob; bi = oi; }
  }
  const float* qr = q_f32 + (size_t)t * cHID;
  const float* ar = a_f32 + (size_t)bi * cHID;
  #pragma unroll
  for (int k = 0; k < 4; ++k) {
    int c = lane + 32 * k;
    rin[(size_t)t * cHID + c] = f2bf(qr[c] + ar[c]);
  }
  if (lane == 0) {
    o_ind[t] = (float)bi;
    float vx = vd[bi * 3 + 0], vy = vd[bi * 3 + 1], vz = vd[bi * 3 + 2];
    o_xyz[t * 3 + 0] = vx; o_xyz[t * 3 + 1] = vy; o_xyz[t * 3 + 2] = vz;
    float ax0 = -vx, ax1 = -vy, ax2 = -vz;
    float ay0 = -ax1, ay1 = ax0, ay2 = 0.f;     // [-ax.y, ax.x, 0]
    float ny = sqrtf(ay0 * ay0 + ay1 * ay1);
    if (ny < 1e-8f) { ay0 = 0.f; ay1 = 1.f; ay2 = 0.f; }
    else { ay0 /= ny; ay1 /= ny; }
    float nx = fmaxf(sqrtf(ax0 * ax0 + ax1 * ax1 + ax2 * ax2), 1e-8f);
    ax0 /= nx; ax1 /= nx; ax2 /= nx;
    float az0 = ax1 * ay2 - ax2 * ay1;
    float az1 = ax2 * ay0 - ax0 * ay2;
    float az2 = ax0 * ay1 - ax1 * ay0;
    float* R = o_rot + (size_t)t * 9;   // columns = axis_x, axis_y, axis_z
    R[0] = ax0; R[1] = ay0; R[2] = az0;
    R[3] = ax1; R[4] = ay1; R[5] = az1;
    R[6] = ax2; R[7] = ay2; R[8] = az2;
  }
}

// ---------------------------------------------------------------------------
extern "C" void kernel_launch(void* const* d_in, const int* in_sizes, int n_in,
                              void* d_out, int out_size, void* d_ws, size_t ws_size,
                              hipStream_t stream) {
  (void)in_sizes; (void)n_in; (void)out_size; (void)ws_size;
  const float* seed  = (const float*)d_in[0];
  const int*   tok   = (const int*)d_in[1];
  const float* Kc    = (const float*)d_in[2];
  const float* dmap  = (const float*)d_in[3];
  const float* dprob = (const float*)d_in[4];
  const float* vdirs = (const float*)d_in[5];
  const float* Wq1 = (const float*)d_in[6];  const float* bq1 = (const float*)d_in[7];
  const float* Wq2 = (const float*)d_in[8];  const float* bq2 = (const float*)d_in[9];
  const float* Wa1 = (const float*)d_in[10]; const float* ba1 = (const float*)d_in[11];
  const float* Wa2 = (const float*)d_in[12]; const float* ba2 = (const float*)d_in[13];
  const float* Wb1 = (const float*)d_in[14]; const float* bb1 = (const float*)d_in[15];
  const float* Wb2 = (const float*)d_in[16]; const float* bb2 = (const float*)d_in[17];
  const float* Wr1 = (const float*)d_in[18]; const float* br1 = (const float*)d_in[19];
  const float* Wr2 = (const float*)d_in[20]; const float* br2 = (const float*)d_in[21];

  float* out   = (float*)d_out;
  float* o_vs  = out;                        // view_score  (2,1024,300)
  float* o_ind = out + cR * cV;              // top_inds    (2,1024)
  float* o_xyz = o_ind + cR;                 // top_xyz     (2,1024,3)
  float* o_rot = o_xyz + cR * 3;             // rot         (2,1024,3,3)
  float* o_res = o_rot + cR * 9;             // res_feat    (2,128,1024)

  char* w = (char*)d_ws;
  auto alloc = [&](size_t bytes) -> void* {
    void* p = (void*)w;
    w += (bytes + 255) & ~(size_t)255;
    return p;
  };
  float*          a_f32  = (float*)alloc((size_t)cV * cHID * 4);
  float*          bias_v = (float*)alloc((size_t)cV * 4);
  unsigned short* ab     = (unsigned short*)alloc((size_t)cHID * cVP * 2);
  unsigned short* qin    = (unsigned short*)alloc((size_t)cR * cKQ * 2);
  unsigned short* Wq1b   = (unsigned short*)alloc((size_t)cKQ * cHID * 2);
  unsigned short* Wq2b   = (unsigned short*)alloc((size_t)cHID * cHID * 2);
  unsigned short* Wr1b   = (unsigned short*)alloc((size_t)cHID * cC * 2);
  unsigned short* Wr2b   = (unsigned short*)alloc((size_t)cC * cC * 2);
  unsigned short* h1     = (unsigned short*)alloc((size_t)cR * cHID * 2);
  float*          q_f32  = (float*)alloc((size_t)cR * cHID * 4);
  unsigned short* q_b    = (unsigned short*)alloc((size_t)cR * cHID * 2);
  unsigned short* rinb   = (unsigned short*)alloc((size_t)cR * cHID * 2);
  unsigned short* h2     = (unsigned short*)alloc((size_t)cR * cHID * 2);

  // 0) weights -> bf16, packed fragment-major
  {
    int total = cKQ * cHID + cHID * cHID + cHID * cC + cC * cC;
    convert_weights_kernel<<<(total + 255) / 256, 256, 0, stream>>>(
        Wq1, Wq2, Wr1, Wr2, Wq1b, Wq2b, Wr1b, Wr2b);
  }
  // 1) anchor MLP + view bias (f32)
  anchor_kernel<<<cV, 128, 0, stream>>>(vdirs, Wa1, ba1, Wa2, ba2,
                                        Wb1, bb1, Wb2, bb2, a_f32, bias_v);
  // 1b) a -> packed aT bf16 (19 n-tiles x 4 k-tiles)
  transpose_a_kernel<<<((cVP / 16) * 4 * 512 + 255) / 256, 256, 0, stream>>>(a_f32, ab);
  // 2) per-token geometry -> q_in (2048 x 160 bf16)
  geom_kernel<<<cR / 8, 256, 0, stream>>>(tok, Kc, dmap, dprob, seed, qin);
  // 3) h1 = relu(q_in @ Wq1 + bq1)
  wmma_gemm_bf16<cKQ, 1><<<dim3(cR / 16, 2), 128, 0, stream>>>(
      qin, cKQ, Wq1b, bq1, nullptr, h1, cHID, 1.f);
  // 4) q = relu(h1 @ Wq2 + bq2)               (f32 + bf16)
  wmma_gemm_bf16<cHID, 1><<<dim3(cR / 16, 2), 128, 0, stream>>>(
      h1, cHID, Wq2b, bq2, q_f32, q_b, cHID, 1.f);
  // 5) view_score = sigmoid(q @ aT / sqrt(128) + bias)  -> d_out
  wmma_gemm_bf16<cHID, 2><<<dim3(cR / 16, 5), 128, 0, stream>>>(
      q_b, cHID, ab, bias_v, o_vs, nullptr, cV, 0.08838834764831845f);
  // 6) argmax + rot + rin = q + a[top]
  select_kernel<<<cR / 8, 256, 0, stream>>>(o_vs, vdirs, a_f32, q_f32,
                                            o_ind, o_xyz, o_rot, rinb);
  // 7) h2 = relu(rin @ Wr1 + br1)
  wmma_gemm_bf16<cHID, 1><<<dim3(cR / 16, 2), 128, 0, stream>>>(
      rinb, cHID, Wr1b, br1, nullptr, h2, cHID, 1.f);
  // 8) res_feat = (h2 @ Wr2 + br2) stored (B,C,M) -> d_out
  wmma_gemm_bf16<cHID, 3><<<dim3(cR / 16, 2), 128, 0, stream>>>(
      h2, cHID, Wr2b, br2, o_res, nullptr, cC, 1.f);
}
